// EfficientSelfAttention_12292196401650
// MI455X (gfx1250) — matrix operand order repaired and checked by reading, not compiled, not run on hardware
//
#include <hip/hip_runtime.h>
#include <hip/hip_bf16.h>
#include <cfloat>

// ---------------------------------------------------------------------------
// EfficientSelfAttention for MI455X (gfx1250, wave32, WMMA)
//   B=2, C=64, RC=8, H=W=96, N=9216
//   QK^T logits : V_WMMA_F32_16X16X4_F32 (K=8 -> 2 wmma/S-tile, fp32,
//                 1/sqrt(N) pre-folded into the Q operand)
//   P*V         : V_WMMA_F32_16X16X32_BF16 (dominant 21.7 GFLOP in bf16)
//   softmax     : pass1 = max only, pass2 = unnormalized exp + denominator,
//                 single normalization in epilogue.
//   blocking    : 1 block = (batch, 32 queries); 4 waves split N 4-way;
//                 each wave keeps 2 q-tiles resident so every K/V tile load
//                 is amortized over 2x the WMMA work (halves L2 traffic).
// ---------------------------------------------------------------------------

typedef __attribute__((ext_vector_type(2)))  float  v2f;
typedef __attribute__((ext_vector_type(8)))  float  v8f;
typedef __attribute__((ext_vector_type(16))) __bf16 v16bf;

#define BATCH 2
#define CH    64
#define RCH   8
#define NN    9216          // 96*96
#define NT16  (NN/16)       // 576 key tiles of 16
#define NT32  (NN/32)       // 288 key tiles of 32
#define QT    (NN/16)       // 576 query tiles of 16
#define CTN   (CH/16)       // 4 channel tiles
#define SCALE (1.0f/96.0f)  // 1/sqrt(N)
#define LDS_STRIDE 34       // dwords per q-row (even -> b64 LDS loads, conflict-free)
#define SL16  (NT16/4)      // 144 16-key tiles per wave slice
#define SL32  (NT32/4)      // 72  32-key tiles per wave slice
#define GQ    2             // q-tiles resident per wave

union U16bf { unsigned u[8]; v16bf v; };

__device__ __forceinline__ unsigned f2bf(float f) {
  unsigned u = __float_as_uint(f);
  return (u + 0x7fffu + ((u >> 16) & 1u)) >> 16;   // RNE
}

// ---------------------------------------------------------------------------
// Kernel A: 1x1-conv projections q,k,v  (one thread per (b,n) column)
// ---------------------------------------------------------------------------
__global__ void __launch_bounds__(256)
proj_kernel(const float* __restrict__ x,
            const float* __restrict__ wq, const float* __restrict__ bq,
            const float* __restrict__ wk, const float* __restrict__ bk,
            const float* __restrict__ wv, const float* __restrict__ bv,
            float* __restrict__ qp,   // (B,N,RC)
            float* __restrict__ kp,   // (B,RC,N)
            float* __restrict__ vp)   // (B,C,N)
{
  int t = blockIdx.x * blockDim.x + threadIdx.x;   // B*N threads
  int b = t / NN, n = t % NN;

  float qa[RCH] = {}, ka[RCH] = {}, va[CH] = {};
  for (int c = 0; c < CH; ++c) {
    float xv = x[((size_t)b * CH + c) * NN + n];
#pragma unroll
    for (int r = 0; r < RCH; ++r) qa[r] += wq[r * CH + c] * xv;
#pragma unroll
    for (int r = 0; r < RCH; ++r) ka[r] += wk[r * CH + c] * xv;
#pragma unroll
    for (int o = 0; o < CH; ++o) va[o] += wv[o * CH + c] * xv;
  }
#pragma unroll
  for (int r = 0; r < RCH; ++r) qp[((size_t)b * NN + n) * RCH + r] = qa[r] + bq[r];
#pragma unroll
  for (int r = 0; r < RCH; ++r) kp[((size_t)b * RCH + r) * NN + n] = ka[r] + bk[r];
#pragma unroll
  for (int o = 0; o < CH; ++o)  vp[((size_t)b * CH + o) * NN + n] = va[o] + bv[o];
}

// ---------------------------------------------------------------------------
// Kernel B1/B2: swizzle Q (B-operand, pre-scaled by 1/sqrt(N)) and K (A-operand)
// into f32 16x16x4 register layout: 4 contiguous dwords per (tile,lane) -> one
// b128 load in the hot loop.   rc = (j>>1)*4 + 2*half + (j&1)
// ---------------------------------------------------------------------------
__global__ void __launch_bounds__(256)
swz_q_kernel(const float* __restrict__ qp, float* __restrict__ qa)
{
  int t = blockIdx.x * blockDim.x + threadIdx.x;   // B*QT*32*4
  int j = t & 3, lane = (t >> 2) & 31;
  int rest = t >> 7;
  int qt = rest % QT, b = rest / QT;
  int half = lane >> 4;
  int rc = (j >> 1) * 4 + 2 * half + (j & 1);
  int q  = qt * 16 + (lane & 15);
  qa[t] = qp[((size_t)b * NN + q) * RCH + rc] * SCALE;   // fold softmax scale
}

__global__ void __launch_bounds__(256)
swz_k_kernel(const float* __restrict__ kp, float* __restrict__ kb)
{
  int t = blockIdx.x * blockDim.x + threadIdx.x;   // B*NT16*32*4
  int j = t & 3, lane = (t >> 2) & 31;
  int rest = t >> 7;
  int nt = rest % NT16, b = rest / NT16;
  int half = lane >> 4;
  int rc = (j >> 1) * 4 + 2 * half + (j & 1);
  int n  = nt * 16 + (lane & 15);
  kb[t] = kp[((size_t)b * RCH + rc) * NN + n];
}

// ---------------------------------------------------------------------------
// Kernel B3: V -> bf16, pre-swizzled into 16-bit A-matrix 16x32 layout
// ---------------------------------------------------------------------------
__global__ void __launch_bounds__(256)
swz_v_kernel(const float* __restrict__ vp, unsigned* __restrict__ va)
{
  int t = blockIdx.x * blockDim.x + threadIdx.x;   // B*CTN*NT32*32*8
  int j = t & 7, lane = (t >> 3) & 31;
  int rest = t >> 8;
  int nt = rest % NT32; rest /= NT32;
  int ct = rest % CTN;  int b = rest / CTN;
  int half = lane >> 4;
  int ch = ct * 16 + (lane & 15);
  int nl = ((j < 4) ? 0 : 16) + 8 * half + 2 * (j & 3);
  int n  = nt * 32 + nl;
  float f0 = vp[((size_t)b * CH + ch) * NN + n];
  float f1 = vp[((size_t)b * CH + ch) * NN + n + 1];
  va[t] = f2bf(f0) | (f2bf(f1) << 16);
}

// ---------------------------------------------------------------------------
// Kernel C: fused attention. 1 block = (batch, 2 q-tiles); 4 waves each own a
// 2304-key slice of N, keep both q-tiles resident, merge through LDS.
// ---------------------------------------------------------------------------
__global__ void __launch_bounds__(128)
attn_kernel(const float* __restrict__ qa, const float* __restrict__ kb,
            const unsigned* __restrict__ va,
            const float* __restrict__ x, const float* __restrict__ gamma,
            float* __restrict__ out)
{
  __shared__ float sm_t[4 * 16 * LDS_STRIDE];  // per-wave P transpose staging
  __shared__ float mred[GQ][4][32];            // per-(qtile,wave) slice max
  __shared__ float lred[GQ][4][32];            // per-(qtile,wave) denominator
  __shared__ float obuf[GQ][CH * 16];          // merged O (2 x 64ch x 16q)

  const int lane = threadIdx.x & 31;
  const int wave = threadIdx.x >> 5;
  const int b    = blockIdx.x / (QT / GQ);
  const int qt0  = (blockIdx.x % (QT / GQ)) * GQ;
  const int half = lane >> 4;
  const int l16  = lane & 15;
  float* sm = &sm_t[wave * 16 * LDS_STRIDE];

  // zero the O merge buffers (completed before the first barrier)
  for (int i = threadIdx.x; i < GQ * CH * 16; i += 128) obuf[0][i] = 0.f;

  // Q operands for both resident q-tiles (pre-scaled by 1/sqrt(N))
  v2f qb1[GQ], qb2[GQ];
#pragma unroll
  for (int u = 0; u < GQ; ++u) {
    float4 qv = *(const float4*)&qa[(((size_t)b * QT + qt0 + u) * 32 + lane) * 4];
    qb1[u] = {qv.x, qv.y};
    qb2[u] = {qv.z, qv.w};
  }

  // S'[n = 16*nt + r + 8*half][q = l16] from one loaded K tile + one Q operand
  auto qk_from_kv = [&](float4 kv, v2f b1, v2f b2) -> v8f {
    v2f a1 = {kv.x, kv.y};
    v2f a2 = {kv.z, kv.w};
    v8f acc = {};
    acc = __builtin_amdgcn_wmma_f32_16x16x4_f32(false, a1, false, b1,
                                                (short)0, acc, false, false);
    acc = __builtin_amdgcn_wmma_f32_16x16x4_f32(false, a2, false, b2,
                                                (short)0, acc, false, false);
    return acc;
  };
  auto load_kv = [&](int nt) -> float4 {
    return *(const float4*)&kb[(((size_t)b * NT16 + nt) * 32 + lane) * 4];
  };

  // ---------- pass 1: per-slice row max (no exp, scale already folded) ------
  float m[GQ];
#pragma unroll
  for (int u = 0; u < GQ; ++u) m[u] = -FLT_MAX;
  for (int i = 0; i < SL16; ++i) {
    float4 kv = load_kv(wave * SL16 + i);          // one K tile, GQ q-tiles
#pragma unroll
    for (int u = 0; u < GQ; ++u) {
      v8f acc = qk_from_kv(kv, qb1[u], qb2[u]);
      float tmax = acc[0];
#pragma unroll
      for (int r = 1; r < 8; ++r) tmax = fmaxf(tmax, acc[r]);
      m[u] = fmaxf(m[u], tmax);
    }
  }
#pragma unroll
  for (int u = 0; u < GQ; ++u) {
    m[u] = fmaxf(m[u], __shfl_xor(m[u], 16));      // combine lane halves
    mred[u][wave][lane] = m[u];
  }
  __syncthreads();
  float M[GQ];
#pragma unroll
  for (int u = 0; u < GQ; ++u)
    M[u] = fmaxf(fmaxf(mred[u][0][lane], mred[u][1][lane]),
                 fmaxf(mred[u][2][lane], mred[u][3][lane]));

  // ---------- pass 2: unnormalized exp, denominator, O = V @ P (bf16) -------
  v8f o[GQ][4] = {};
  float Lp[GQ] = {};
  for (int i = 0; i < SL32; ++i) {
    const int nt2 = wave * SL32 + i;
    float4 kva = load_kv(nt2 * 2);
    float4 kvb = load_kv(nt2 * 2 + 1);

    // build bf16 P operand for each resident q-tile via LDS transpose
    U16bf pb[GQ];
#pragma unroll
    for (int u = 0; u < GQ; ++u) {
      v8f s0 = qk_from_kv(kva, qb1[u], qb2[u]);
      v8f s1 = qk_from_kv(kvb, qb1[u], qb2[u]);
#pragma unroll
      for (int r = 0; r < 8; ++r) {
        float p0 = __expf(s0[r] - M[u]);           // <= 1, unnormalized
        float p1 = __expf(s1[r] - M[u]);
        Lp[u] += p0 + p1;
        sm[l16 * LDS_STRIDE +      8 * half + r] = p0;
        sm[l16 * LDS_STRIDE + 16 + 8 * half + r] = p1;
      }
      asm volatile("s_wait_dscnt 0" ::: "memory"); // same-wave LDS RAW fence
#pragma unroll
      for (int j = 0; j < 8; ++j) {
        float2 pv2 = *(const float2*)&sm[l16 * LDS_STRIDE + (j >> 2) * 16 + 8 * half + 2 * (j & 3)];
        pb[u].u[j] = f2bf(pv2.x) | (f2bf(pv2.y) << 16);
      }
    }

    // one V tile load feeds GQ q-tiles of PV accumulation
#pragma unroll
    for (int ct = 0; ct < 4; ++ct) {
      const uint4* vp4 =
          (const uint4*)&va[((((size_t)b * CTN + ct) * NT32 + nt2) * 32 + lane) * 8];
      uint4 u0 = vp4[0], u1 = vp4[1];
      U16bf av;
      av.u[0] = u0.x; av.u[1] = u0.y; av.u[2] = u0.z; av.u[3] = u0.w;
      av.u[4] = u1.x; av.u[5] = u1.y; av.u[6] = u1.z; av.u[7] = u1.w;
#pragma unroll
      for (int u = 0; u < GQ; ++u)
        o[u][ct] = __builtin_amdgcn_wmma_f32_16x16x32_bf16(
            false, av.v, false, pb[u].v, (short)0, o[u][ct], false, false);
    }
  }

  // ---------- merge the 4 slices ----------
#pragma unroll
  for (int u = 0; u < GQ; ++u) {
    Lp[u] += __shfl_xor(Lp[u], 16);    // halves cover disjoint keys
    lred[u][wave][lane] = Lp[u];
#pragma unroll
    for (int ct = 0; ct < 4; ++ct) {
#pragma unroll
      for (int r = 0; r < 8; ++r) {
        int ch = ct * 16 + 8 * half + r;
        atomicAdd(&obuf[u][ch * 16 + l16], o[u][ct][r]);   // ds_add_f32
      }
    }
  }
  __syncthreads();

  // ---------- epilogue: out = x + gamma * O / L ----------
  const float g = gamma[0];
  const int q  = threadIdx.x & 15;
  const int cb = (threadIdx.x >> 4) * 8;
#pragma unroll
  for (int u = 0; u < GQ; ++u) {
    const float Lq = lred[u][0][q] + lred[u][1][q] + lred[u][2][q] + lred[u][3][q];
    const float gi = g / Lq;
    const int nq = (qt0 + u) * 16 + q;
#pragma unroll
    for (int i = 0; i < 8; ++i) {
      int ch = cb + i;
      size_t idx = ((size_t)b * CH + ch) * NN + nq;
      out[idx] = x[idx] + gi * obuf[u][ch * 16 + q];
    }
  }
}

// ---------------------------------------------------------------------------
extern "C" void kernel_launch(void* const* d_in, const int* in_sizes, int n_in,
                              void* d_out, int out_size, void* d_ws, size_t ws_size,
                              hipStream_t stream)
{
  (void)in_sizes; (void)n_in; (void)out_size; (void)ws_size;
  const float* x  = (const float*)d_in[0];
  const float* wq = (const float*)d_in[1];
  const float* bq = (const float*)d_in[2];
  const float* wk = (const float*)d_in[3];
  const float* bk = (const float*)d_in[4];
  const float* wv = (const float*)d_in[5];
  const float* bv = (const float*)d_in[6];
  const float* gm = (const float*)d_in[7];
  float* out = (float*)d_out;

  // workspace carve-up (~9.4 MB total, all 16B-aligned)
  float* qp  = (float*)d_ws;                           // B*N*RC
  float* kp  = qp  + (size_t)BATCH * NN * RCH;         // B*RC*N
  float* vp  = kp  + (size_t)BATCH * RCH * NN;         // B*C*N
  float* qaw = vp  + (size_t)BATCH * CH * NN;          // B*QT*32*4
  float* kbw = qaw + (size_t)BATCH * QT * 32 * 4;      // B*NT16*32*4
  unsigned* vaw = (unsigned*)(kbw + (size_t)BATCH * NT16 * 32 * 4); // B*CT*NT32*32*8

  proj_kernel <<<(BATCH * NN) / 256, 256, 0, stream>>>(x, wq, bq, wk, bk, wv, bv,
                                                       qp, kp, vp);
  swz_q_kernel<<<(BATCH * QT  * 32 * 4) / 256, 256, 0, stream>>>(qp, qaw);
  swz_k_kernel<<<(BATCH * NT16 * 32 * 4) / 256, 256, 0, stream>>>(kp, kbw);
  swz_v_kernel<<<(BATCH * CTN * NT32 * 32 * 8) / 256, 256, 0, stream>>>(vp, vaw);
  attn_kernel <<<BATCH * (QT / GQ), 128, 0, stream>>>(qaw, kbw, vaw, x, gm, out);
}
